// Block_47193100648803
// MI455X (gfx1250) — compile-verified, matrix operand order checked by reading, not run on hardware
//
#include <hip/hip_runtime.h>

// ---------------------------------------------------------------------------
// MI455X (gfx1250) implementation of the spatio-temporal attention block.
// All large GEMMs run on v_wmma_f32_16x16x32_f16 (wave32 WMMA), with fused
// bias / exact-GELU / residual / f16-downconvert epilogues so intermediate
// traffic is halved and pass count minimized (~90 GFLOP of GEMM vs ~1.4 GB of
// activation traffic at 23.3 TB/s -> data-movement bound; fuse everything).
// Tile staging: A-operand via gfx1250 async global->LDS (ASYNCcnt); B-operand
// (weight panel) via the Tensor Data Mover (TENSORcnt), using the TDM LDS
// padding feature to produce the bank-conflict-free 40-half row stride.
// ---------------------------------------------------------------------------

typedef __attribute__((ext_vector_type(8)))  _Float16 v8h;
typedef __attribute__((ext_vector_type(16))) _Float16 v16h;
typedef __attribute__((ext_vector_type(8)))  float    v8f;
typedef __attribute__((ext_vector_type(4)))  unsigned v4u;
typedef __attribute__((ext_vector_type(8)))  int      v8i_t;
typedef __attribute__((ext_vector_type(4)))  int      v4i_t;

namespace cfg {
constexpr int N  = 8, C = 256, T = 128, V = 25;
constexpr int H  = 8, HC = 32, W = 3, O = 256, PADT = 1;
constexpr int B  = N * T;        // 1024
constexpr int L  = W * V;        // 75
constexpr int M  = B * L;        // 76800  (rows of the flat sequence)
constexpr int M2 = N * T * V;    // 25600  (rows of the temporal-conv GEMM)
constexpr int K2 = W * C;        // 768
}

union V16U { v16h v; struct { v8h lo, hi; } p; };

// Fragment load per documented 16-bit WMMA operand layout: each lane holds two
// contiguous 8-element K chunks 16 apart (lanes<16: K 0-7 & 16-23 of its row).
__device__ __forceinline__ v16h frag_ld(const _Float16* p) {
    V16U u;
    u.p.lo = *(const v8h*)p;
    u.p.hi = *(const v8h*)(p + 16);
    return u.v;
}

__device__ __forceinline__ float gelu_exact(float x) {
    return 0.5f * x * (1.0f + erff(x * 0.70710678118654752f));
}

// gfx1250 async global->LDS copy (16B per lane), tracked by ASYNCcnt.
// GVS addressing: mem = SGPR base + per-lane 32-bit byte offset.
__device__ __forceinline__ void async_ld_b128(unsigned lds_addr, unsigned goff,
                                              const void* sbase) {
    asm volatile("global_load_async_to_lds_b128 %0, %1, %2"
                 :
                 : "v"(lds_addr), "v"(goff), "s"(sbase)
                 : "memory");
}
__device__ __forceinline__ void wait_async0() {
    asm volatile("s_wait_asynccnt 0" ::: "memory");
}
// Low 32 bits of a generic pointer into LDS == wave-relative LDS byte address.
__device__ __forceinline__ unsigned lds_addr_of(const void* p) {
    return (unsigned)(size_t)p;
}

// ---------------------------------------------------------------------------
// TDM: DMA one 128-row x 64-byte weight tile (row stride K*2 bytes) into LDS.
// data_size = 8B units; tile_dim0 = 8 (64B), tile_dim1 = 128 rows.
// LDS padding: after every 16 DWORDs (one 64B row) insert 4 DWORDs (16B) so
// rows land with the 40-half stride the ds_load_b128 fragment reads expect.
// D# packing per CDNA5 ISA 8.3/8.4; groups 2/3 zero (tile_dim2=0 => unused).
// This toolchain's builtin takes 6 args (extra int32x8 group before cpol).
// ---------------------------------------------------------------------------
__device__ __forceinline__ void tdm_load_b_tile(unsigned lds, unsigned long long gaddr,
                                                int K) {
    union G0U { unsigned long long q[2]; v4u v; } g0;
    union G1U { unsigned long long q[4]; v8i_t v; } g1;

    g0.q[0] = 1ull                                   // count=1 (valid)
            | ((unsigned long long)lds << 32);       // lds_addr
    g0.q[1] = (gaddr & 0x1FFFFFFFFFFFFFFull)         // global_addr[56:0]
            | (2ull << 62);                          // type = 2 ("image")

    const unsigned long long td0     = (unsigned long long)(K >> 2); // row len, 8B units
    const unsigned long long stride0 = (unsigned long long)(K >> 2); // row stride, 8B units
    const unsigned long long flags   = (3ull << 16)   // data_size = 8B
                                     | (1ull << 20)   // pad_enable
                                     | (3ull << 22)   // pad_interval = 16 DWORDs
                                     | (3ull << 25);  // pad_amount   = 4 DWORDs
    g1.q[0] = flags
            | ((td0 & 0xFFFFull) << 48);             // tensor_dim0[15:0]
    g1.q[1] = ((td0 >> 16) & 0xFFFFull)              // tensor_dim0[31:16]
            | (128ull << 16)                         // tensor_dim1 = 128
            | (8ull << 48);                          // tile_dim0 = 8 (64B)
    g1.q[2] = 128ull                                 // tile_dim1 = 128
            | ((stride0 & 0xFFFFFFFFull) << 32);     // tensor_dim0_stride[31:0]
    g1.q[3] = (stride0 >> 32) & 0xFFFFull;           // stride[47:32]; dim1_stride=0

    const v4i_t z4 = {0, 0, 0, 0};
    const v8i_t z8 = {0, 0, 0, 0, 0, 0, 0, 0};
    __builtin_amdgcn_tensor_load_to_lds(g0.v, g1.v, z4, z4, z8, 0);
}

// ---------------------------------------------------------------------------
// Tiled WMMA GEMM:  out[M x 256] = A[M x K] (f16) * BT^T  (BT stored [256][K])
// Block tile 128x128, K-step 32, 256 threads = 8 waves (4 along M, 2 along N),
// each wave computes 32x64 via 2x4 accumulators of v_wmma_f32_16x16x32_f16.
// Double-buffered LDS: A tiles via async global->LDS, B tiles via TDM.
// ---------------------------------------------------------------------------
template<bool GELU, bool RES, bool OUT32, bool OUT16, bool GATHER, bool PERM>
__global__ void __launch_bounds__(256)
gemm_wmma(const _Float16* __restrict__ A, const _Float16* __restrict__ BT,
          const float* __restrict__ bias, const float* __restrict__ res,
          float* __restrict__ out32, _Float16* __restrict__ out16,
          int Mrows, int K) {
    using namespace cfg;
    __shared__ __align__(16) _Float16 As[2][128][40];
    __shared__ __align__(16) _Float16 Bs[2][128][40];

    const int tid  = threadIdx.x;
    const int lane = tid & 31;
    const int wid  = tid >> 5;
    const int wm   = wid & 3;     // 4 waves along M (each 32 rows)
    const int wn   = wid >> 2;    // 2 waves along N (each 64 cols)
    const int lrow = tid >> 1;          // 0..127 : tile row this thread loads
    const int lcol = (tid & 1) * 16;    // 0 or 16: which half of the 32-K row
    const int half = lane >> 4;
    const int r16  = lane & 15;
    const int ka   = half * 8;          // per-lane K-chunk base in LDS row

    const int gmA = blockIdx.x * 128 + lrow;   // A tile global row

    // Byte offsets from the uniform A base pointer (all buffers < 2 GiB).
    auto aOff = [&](int kt) -> unsigned {
        int gk = kt * 32 + lcol;
        if (!GATHER) {
            return (unsigned)(((size_t)gmA * K + gk) * sizeof(_Float16));
        } else {
            // A row (n,t,v) gathers window w from h[((n*T+t)*W+w)*V+v, i]
            int nt = gmA / V, v = gmA - nt * V;
            int w  = gk >> 8, i = gk & 255;
            return (unsigned)((((size_t)((nt * W + w) * V + v)) * C + i) *
                              sizeof(_Float16));
        }
    };

    // A tile: two async 16B copies (32B of one row half) per thread.
    auto issue_async_A = [&](int kt, int bufi) {
        const unsigned ao = aOff(kt);
        const unsigned la = lds_addr_of(&As[bufi][lrow][lcol]);
        async_ld_b128(la,      ao,      A);
        async_ld_b128(la + 16, ao + 16, A);
    };
    // B tile: one TDM descriptor, issued by wave 0 only.
    const unsigned long long bBase =
        (unsigned long long)(size_t)BT + (size_t)(blockIdx.y * 128) * K * sizeof(_Float16);
    auto issue_tdm_B = [&](int kt, int bufi) {
        if (wid == 0)
            tdm_load_b_tile(lds_addr_of(&Bs[bufi][0][0]),
                            bBase + (unsigned long long)kt * 64ull, K);
    };

    v8f acc[2][4] = {};
    const int KT = K / 32;

    issue_async_A(0, 0);
    issue_tdm_B(0, 0);
    wait_async0();
    if (wid == 0) __builtin_amdgcn_s_wait_tensorcnt(0);
    __syncthreads();

    for (int kt = 0; kt < KT; ++kt) {
        const int cur = kt & 1;
        if (kt + 1 < KT) {
            issue_async_A(kt + 1, cur ^ 1);
            issue_tdm_B(kt + 1, cur ^ 1);
        }
        if (kt + 2 < KT)                // L2 prefetch two steps ahead (A stream)
            __builtin_prefetch((const char*)A + aOff(kt + 2), 0, 0);

        v16h af[2], bf[4];
        #pragma unroll
        for (int tm = 0; tm < 2; ++tm)
            af[tm] = frag_ld(&As[cur][wm * 32 + tm * 16 + r16][ka]);
        #pragma unroll
        for (int tn = 0; tn < 4; ++tn)
            bf[tn] = frag_ld(&Bs[cur][wn * 64 + tn * 16 + r16][ka]);

        #pragma unroll
        for (int tm = 0; tm < 2; ++tm)
            #pragma unroll
            for (int tn = 0; tn < 4; ++tn)
                acc[tm][tn] = __builtin_amdgcn_wmma_f32_16x16x32_f16(
                    false, af[tm], false, bf[tn], (short)0, acc[tm][tn],
                    false, false);

        if (kt + 1 < KT) {
            wait_async0();
            if (wid == 0) __builtin_amdgcn_s_wait_tensorcnt(0);
        }
        __syncthreads();
    }

    // Epilogue. C/D layout: VGPR r holds M=r (lanes 0-15) / M=r+8 (lanes 16-31),
    // N = lane&15.
    #pragma unroll
    for (int tm = 0; tm < 2; ++tm) {
        const int gm_base = blockIdx.x * 128 + wm * 32 + tm * 16 + half * 8;
        #pragma unroll
        for (int tn = 0; tn < 4; ++tn) {
            const int gn = blockIdx.y * 128 + wn * 64 + tn * 16 + r16;
            const float bs = bias[gn];
            #pragma unroll
            for (int r8 = 0; r8 < 8; ++r8) {
                const int gm = gm_base + r8;
                float v = acc[tm][tn][r8] + bs;
                if (GELU) v = gelu_exact(v);
                if (RES)  v += res[(size_t)gm * 256 + gn];
                if (PERM) {
                    // rows are (n,t,v); store to (N, O, T, V)
                    int n_  = gm / (T * V);
                    int rem = gm - n_ * (T * V);
                    int t_  = rem / V;
                    int v_  = rem - t_ * V;
                    out32[(((size_t)n_ * O + gn) * T + t_) * V + v_] = v;
                } else {
                    if (OUT32) out32[(size_t)gm * 256 + gn] = v;
                    if (OUT16) out16[(size_t)gm * 256 + gn] = (_Float16)v;
                }
            }
        }
    }
    (void)Mrows;
}

// ---------------------------------------------------------------------------
// Fused unfold x(N,C,T,V) -> seq(M,C) + LayerNorm -> nx(f16).
// One block per flat row; saves re-reading seq for LN (78 MB pass).
// ---------------------------------------------------------------------------
__global__ void __launch_bounds__(256)
unfold_ln_kernel(const float* __restrict__ x, const float* __restrict__ g,
                 const float* __restrict__ bta, float* __restrict__ seq,
                 _Float16* __restrict__ nx) {
    using namespace cfg;
    __shared__ float red[256];
    const int row = blockIdx.x;     // 0..M-1
    const int c   = threadIdx.x;    // 0..255
    const int b = row / L, l = row - b * L;
    const int w = l / V,  v = l - w * V;
    const int n = b / T,  t = b - n * T;
    const int tw = t + w - PADT;
    float val = 0.0f;
    if (tw >= 0 && tw < T)
        val = x[(((size_t)n * C + c) * T + tw) * V + v];
    seq[(size_t)row * C + c] = val;

    red[c] = val; __syncthreads();
    #pragma unroll
    for (int s = 128; s > 0; s >>= 1) { if (c < s) red[c] += red[c + s]; __syncthreads(); }
    const float mean = red[0] * (1.0f / C);
    __syncthreads();
    const float d = val - mean;
    red[c] = d * d; __syncthreads();
    #pragma unroll
    for (int s = 128; s > 0; s >>= 1) { if (c < s) red[c] += red[c + s]; __syncthreads(); }
    const float rinv = rsqrtf(red[0] * (1.0f / C) + 1e-5f);
    nx[(size_t)row * C + c] = (_Float16)(d * rinv * g[c] + bta[c]);
}

// ---------------------------------------------------------------------------
// LayerNorm over C=256 per row, fp32 in -> f16 out (feeds next WMMA GEMM).
// ---------------------------------------------------------------------------
__global__ void __launch_bounds__(256)
ln16_kernel(const float* __restrict__ in, const float* __restrict__ g,
            const float* __restrict__ bta, _Float16* __restrict__ out) {
    using namespace cfg;
    __shared__ float red[256];
    const int row = blockIdx.x;
    const int c   = threadIdx.x;
    const float v = in[(size_t)row * C + c];
    red[c] = v; __syncthreads();
    #pragma unroll
    for (int s = 128; s > 0; s >>= 1) { if (c < s) red[c] += red[c + s]; __syncthreads(); }
    const float mean = red[0] * (1.0f / C);
    __syncthreads();
    const float d = v - mean;
    red[c] = d * d; __syncthreads();
    #pragma unroll
    for (int s = 128; s > 0; s >>= 1) { if (c < s) red[c] += red[c + s]; __syncthreads(); }
    const float rinv = rsqrtf(red[0] * (1.0f / C) + 1e-5f);
    out[(size_t)row * C + c] = (_Float16)(d * rinv * g[c] + bta[c]);
}

// ---------------------------------------------------------------------------
// Pooled-query attention: per (b): logits (L,H) -> softmax over L -> pq(B,C).
// One block per b; wave w handles head h=w (wave32 shuffle reductions).
// ---------------------------------------------------------------------------
__global__ void __launch_bounds__(256)
pool_q_kernel(const float* __restrict__ q, const float* __restrict__ Wqa,
              const float* __restrict__ bqa, float* __restrict__ pq) {
    using namespace cfg;
    __shared__ float sw[H][L];
    const int b    = blockIdx.x;
    const int h    = threadIdx.x >> 5;
    const int lane = threadIdx.x & 31;
    const float scale = 0.17677669529663687f;  // 1/sqrt(HC)

    for (int l = 0; l < L; ++l) {
        const float* qr = q + ((size_t)b * L + l) * C;
        const int c0 = lane * 8;
        float p = 0.0f;
        #pragma unroll
        for (int j = 0; j < 8; ++j) p += qr[c0 + j] * Wqa[(c0 + j) * H + h];
        #pragma unroll
        for (int off = 16; off > 0; off >>= 1) p += __shfl_xor(p, off, 32);
        if (lane == 0) sw[h][l] = (p + bqa[h]) * scale;
    }
    __syncthreads();

    float mx = -1e30f;
    for (int l = lane; l < L; l += 32) mx = fmaxf(mx, sw[h][l]);
    #pragma unroll
    for (int off = 16; off > 0; off >>= 1) mx = fmaxf(mx, __shfl_xor(mx, off, 32));
    float sum = 0.0f;
    for (int l = lane; l < L; l += 32) { float e = __expf(sw[h][l] - mx); sw[h][l] = e; sum += e; }
    #pragma unroll
    for (int off = 16; off > 0; off >>= 1) sum += __shfl_xor(sum, off, 32);
    const float inv = 1.0f / sum;
    __syncthreads();

    float acc = 0.0f;
    for (int l = 0; l < L; ++l)
        acc += sw[h][l] * inv * q[((size_t)b * L + l) * C + h * HC + lane];
    pq[(size_t)b * C + h * HC + lane] = acc;
}

__global__ void __launch_bounds__(256)
pool_k_kernel(const float* __restrict__ keys, const float* __restrict__ pq,
              const float* __restrict__ Wka, const float* __restrict__ bka,
              float* __restrict__ pk) {
    using namespace cfg;
    __shared__ float sw[H][L];
    const int b    = blockIdx.x;
    const int h    = threadIdx.x >> 5;
    const int lane = threadIdx.x & 31;
    const float scale = 0.17677669529663687f;

    for (int l = 0; l < L; ++l) {
        const float* kr = keys + ((size_t)b * L + l) * C;
        const int c0 = lane * 8;
        float p = 0.0f;
        #pragma unroll
        for (int j = 0; j < 8; ++j)
            p += kr[c0 + j] * pq[(size_t)b * C + c0 + j] * Wka[(c0 + j) * H + h];
        #pragma unroll
        for (int off = 16; off > 0; off >>= 1) p += __shfl_xor(p, off, 32);
        if (lane == 0) sw[h][l] = (p + bka[h]) * scale;
    }
    __syncthreads();

    float mx = -1e30f;
    for (int l = lane; l < L; l += 32) mx = fmaxf(mx, sw[h][l]);
    #pragma unroll
    for (int off = 16; off > 0; off >>= 1) mx = fmaxf(mx, __shfl_xor(mx, off, 32));
    float sum = 0.0f;
    for (int l = lane; l < L; l += 32) { float e = __expf(sw[h][l] - mx); sw[h][l] = e; sum += e; }
    #pragma unroll
    for (int off = 16; off > 0; off >>= 1) sum += __shfl_xor(sum, off, 32);
    const float inv = 1.0f / sum;
    __syncthreads();

    float acc = 0.0f;
    for (int l = 0; l < L; ++l)
        acc += sw[h][l] * inv * keys[((size_t)b * L + l) * C + h * HC + lane];
    pk[(size_t)b * C + h * HC + lane] = acc;
}

// vals * pk (broadcast per b) -> f16 A-operand of the Wt GEMM
__global__ void __launch_bounds__(256)
scale_vals_kernel(const float* __restrict__ vals, const float* __restrict__ pk,
                  _Float16* __restrict__ out) {
    using namespace cfg;
    const size_t idx = (size_t)blockIdx.x * 256 + threadIdx.x;
    const int    c   = (int)(idx & 255);
    const size_t m   = idx >> 8;
    const int    b   = (int)(m / L);
    out[idx] = (_Float16)(vals[idx] * pk[(size_t)b * C + c]);
}

// (kv + q) -> f16 A-operand of the Wp GEMM
__global__ void __launch_bounds__(256)
add2_f16_kernel(const float* __restrict__ a, const float* __restrict__ b,
                _Float16* __restrict__ out) {
    const size_t idx = (size_t)blockIdx.x * 256 + threadIdx.x;
    out[idx] = (_Float16)(a[idx] + b[idx]);
}

// W[K][256] fp32 -> BT[256][K] f16 (N-major so B-tiles load contiguously)
__global__ void __launch_bounds__(256)
wconvT_kernel(const float* __restrict__ Wsrc, _Float16* __restrict__ BT, int K) {
    const int idx = blockIdx.x * 256 + threadIdx.x;  // K*256 total
    const int k = idx >> 8, n = idx & 255;
    BT[(size_t)n * K + k] = (_Float16)Wsrc[idx];
}

// ---------------------------------------------------------------------------
extern "C" void kernel_launch(void* const* d_in, const int* in_sizes, int n_in,
                              void* d_out, int out_size, void* d_ws, size_t ws_size,
                              hipStream_t stream) {
    using namespace cfg;
    (void)in_sizes; (void)n_in; (void)out_size; (void)ws_size;

    const float* x     = (const float*)d_in[0];
    const float* ln1_g = (const float*)d_in[1];
    const float* ln1_b = (const float*)d_in[2];
    const float* Wq    = (const float*)d_in[3];
    const float* bq    = (const float*)d_in[4];
    const float* Wqa   = (const float*)d_in[5];
    const float* bqa   = (const float*)d_in[6];
    const float* Wk    = (const float*)d_in[7];
    const float* bk    = (const float*)d_in[8];
    const float* Wka   = (const float*)d_in[9];
    const float* bka   = (const float*)d_in[10];
    const float* Wv    = (const float*)d_in[11];
    const float* bv    = (const float*)d_in[12];
    const float* Wt    = (const float*)d_in[13];
    const float* bt    = (const float*)d_in[14];
    const float* Wp    = (const float*)d_in[15];
    const float* bp    = (const float*)d_in[16];
    const float* ffn_g = (const float*)d_in[17];
    const float* ffn_b = (const float*)d_in[18];
    const float* W1    = (const float*)d_in[19];
    const float* b1    = (const float*)d_in[20];
    const float* W2    = (const float*)d_in[21];
    const float* b2    = (const float*)d_in[22];
    const float* tn_g  = (const float*)d_in[23];
    const float* tn_b  = (const float*)d_in[24];
    const float* c1_w  = (const float*)d_in[25];
    const float* c1_b  = (const float*)d_in[26];
    const float* c2_w  = (const float*)d_in[27];
    const float* c2_b  = (const float*)d_in[28];

    // ---- workspace carve-up (aliased through the pipeline) ----
    char* p = (char*)d_ws;
    auto alloc = [&](size_t bytes) -> void* {
        void* r = (void*)p;
        p += (bytes + 255) & ~(size_t)255;
        return r;
    };
    const size_t szMCf = (size_t)M * C * sizeof(float);
    const size_t szMCh = (size_t)M * C * sizeof(_Float16);

    float*    seq  = (float*)alloc(szMCf);          // seq  -> later y
    float*    qbuf = (float*)alloc(szMCf);          // q    -> later att
    float*    keys = (float*)alloc(szMCf);          // keys -> later kv
    float*    vals = (float*)alloc(szMCf);          // vals
    _Float16* f16a = (_Float16*)alloc(szMCh);       // nx -> aq -> fln -> t5
    _Float16* f16b = (_Float16*)alloc(szMCh);       // vs -> g1 -> h
    float*    pq   = (float*)alloc((size_t)B * C * sizeof(float));
    float*    pk   = (float*)alloc((size_t)B * C * sizeof(float));
    _Float16* wq16 = (_Float16*)alloc((size_t)C * C * sizeof(_Float16));
    _Float16* wk16 = (_Float16*)alloc((size_t)C * C * sizeof(_Float16));
    _Float16* wv16 = (_Float16*)alloc((size_t)C * C * sizeof(_Float16));
    _Float16* wt16 = (_Float16*)alloc((size_t)C * C * sizeof(_Float16));
    _Float16* wp16 = (_Float16*)alloc((size_t)C * C * sizeof(_Float16));
    _Float16* w116 = (_Float16*)alloc((size_t)C * C * sizeof(_Float16));
    _Float16* w216 = (_Float16*)alloc((size_t)C * C * sizeof(_Float16));
    _Float16* c116 = (_Float16*)alloc((size_t)C * C * sizeof(_Float16));
    _Float16* c2T  = (_Float16*)alloc((size_t)K2 * O * sizeof(_Float16));

    float*    att = qbuf;   // aliases (stage-disjoint lifetimes)
    float*    kv  = keys;
    float*    y   = seq;
    _Float16* nx  = f16a;

    // ---- weight convert + transpose to f16 N-major ----
    wconvT_kernel<<<C, 256, 0, stream>>>(Wq,   wq16, C);
    wconvT_kernel<<<C, 256, 0, stream>>>(Wk,   wk16, C);
    wconvT_kernel<<<C, 256, 0, stream>>>(Wv,   wv16, C);
    wconvT_kernel<<<C, 256, 0, stream>>>(Wt,   wt16, C);
    wconvT_kernel<<<C, 256, 0, stream>>>(Wp,   wp16, C);
    wconvT_kernel<<<C, 256, 0, stream>>>(W1,   w116, C);
    wconvT_kernel<<<C, 256, 0, stream>>>(W2,   w216, C);
    wconvT_kernel<<<C, 256, 0, stream>>>(c1_w, c116, C);
    wconvT_kernel<<<K2, 256, 0, stream>>>(c2_w, c2T, K2);   // (w,i,o) flat == [k][o]

    const dim3 gemmGrid(M / 128, 2, 1);
    const dim3 gemmGrid2(M2 / 128, 2, 1);

    // ---- unfold + LN1 (fused) ----
    unfold_ln_kernel<<<M, 256, 0, stream>>>(x, ln1_g, ln1_b, seq, nx);

    // ---- q / keys / vals projections (WMMA) ----
    gemm_wmma<false,false,true,false,false,false><<<gemmGrid, 256, 0, stream>>>(
        nx, wq16, bq, nullptr, qbuf, nullptr, M, C);
    pool_q_kernel<<<B, 256, 0, stream>>>(qbuf, Wqa, bqa, pq);

    gemm_wmma<false,false,true,false,false,false><<<gemmGrid, 256, 0, stream>>>(
        nx, wk16, bk, nullptr, keys, nullptr, M, C);
    pool_k_kernel<<<B, 256, 0, stream>>>(keys, pq, Wka, bka, pk);

    gemm_wmma<false,false,true,false,false,false><<<gemmGrid, 256, 0, stream>>>(
        nx, wv16, bv, nullptr, vals, nullptr, M, C);

    // ---- kv = (vals * pk) @ Wt + bt ----
    scale_vals_kernel<<<M, 256, 0, stream>>>(vals, pk, f16b);
    gemm_wmma<false,false,true,false,false,false><<<gemmGrid, 256, 0, stream>>>(
        f16b, wt16, bt, nullptr, kv, nullptr, M, C);

    // ---- att = (kv + q) @ Wp + bp + seq  (residual fused) ----
    add2_f16_kernel<<<M, 256, 0, stream>>>(kv, qbuf, f16a);
    gemm_wmma<false,true,true,false,false,false><<<gemmGrid, 256, 0, stream>>>(
        f16a, wp16, bp, seq, att, nullptr, M, C);

    // ---- FFN: y = gelu(LN(att) @ W1 + b1) @ W2 + b2 + att ----
    ln16_kernel<<<M, 256, 0, stream>>>(att, ffn_g, ffn_b, f16a);
    gemm_wmma<true,false,false,true,false,false><<<gemmGrid, 256, 0, stream>>>(
        f16a, w116, b1, nullptr, nullptr, f16b, M, C);
    gemm_wmma<false,true,true,false,false,false><<<gemmGrid, 256, 0, stream>>>(
        f16b, w216, b2, att, y, nullptr, M, C);

    // ---- temporal branch: h = gelu(LN(y) @ c1_w + c1_b) ----
    ln16_kernel<<<M, 256, 0, stream>>>(y, tn_g, tn_b, f16a);
    gemm_wmma<true,false,false,true,false,false><<<gemmGrid, 256, 0, stream>>>(
        f16a, c116, c1_b, nullptr, nullptr, f16b, M, C);

    // ---- out[n,t,v,o] = sum_{w,i} h[(nt*W+w)*V+v, i] * c2_w[w,i,o] + c2_b
    //      gathered-A GEMM, K=768, permuted store to (N, O, T, V) ----
    gemm_wmma<false,false,true,false,true,true><<<gemmGrid2, 256, 0, stream>>>(
        f16b, c2T, c2_b, nullptr, (float*)d_out, nullptr, M2, K2);
}
